// Attn_20246475834088
// MI455X (gfx1250) — matrix operand order repaired and checked by reading, not compiled, not run on hardware
//
#include <hip/hip_runtime.h>
#include <type_traits>

// ---------------- problem constants ----------------
#define BB 2
#define TT 2048
#define DD 1024
#define NHH 32
#define HDD 32            // head dim == wave32 lanes
#define BT (BB * TT)      // 4096 rows
#define QK_SCALE 0.17677669529663689f  // 1/sqrt(32)

typedef unsigned int u32;
typedef __attribute__((ext_vector_type(16))) _Float16 v16h;
typedef __attribute__((ext_vector_type(8)))  float    v8f;
typedef __attribute__((ext_vector_type(4)))  _Float16 h4;
typedef __attribute__((ext_vector_type(4)))  float    f4;

union F16x16 { u32 u[8]; v16h v; };

// =====================================================================
// Tiled GEMM: C[M,N] = A[M,K] * Bw[N,K]^T   (fp32 accum, f16 WMMA)
// block: 256 threads = 8 waves; tile BM=128 x BN=64 x BK=32
// wave w computes rows [w*16, w*16+16) x all 64 cols (4 wmma accums)
// =====================================================================
template <typename TA>
__global__ void gemm_xwT(const TA* __restrict__ A, const float* __restrict__ Bw,
                         float* __restrict__ C, int M, int N, int K) {
  __shared__ alignas(16) _Float16 sA[128 * 32];
  __shared__ alignas(16) _Float16 sB[64 * 32];
  const int tid  = threadIdx.x;
  const int wid  = tid >> 5;
  const int lane = tid & 31;
  const int hs   = lane >> 4;     // lane half (K-half select)
  const int nn   = lane & 15;     // M (A-frag) / N (B-frag / C cols)
  const int m0   = blockIdx.x * 128;
  const int n0   = blockIdx.y * 64;

  const v8f vz = {0, 0, 0, 0, 0, 0, 0, 0};
  v8f acc[4];
#pragma unroll
  for (int i = 0; i < 4; ++i) acc[i] = vz;

  const int nkt = K / 32;
  for (int kt = 0; kt < nkt; ++kt) {
    const int k0 = kt * 32;
    __syncthreads();
    // ---- stage A tile (128x32) fp32/f16 -> f16 LDS ----
#pragma unroll
    for (int c = 0; c < 4; ++c) {
      int idx = tid + c * 256;
      int row = idx >> 3, ch = idx & 7;
      const TA* src = A + (size_t)(m0 + row) * K + k0 + ch * 4;
      h4 hv;
      if constexpr (std::is_same<TA, float>::value) {
        f4 a = *(const f4*)src;
        hv = (h4){(_Float16)a[0], (_Float16)a[1], (_Float16)a[2], (_Float16)a[3]};
      } else {
        hv = *(const h4*)src;
      }
      *(h4*)&sA[row * 32 + ch * 4] = hv;
    }
    // ---- stage B tile (64x32) fp32 -> f16 LDS ----
#pragma unroll
    for (int c = 0; c < 2; ++c) {
      int idx = tid + c * 256;
      int row = idx >> 3, ch = idx & 7;
      f4 bv = *(const f4*)(Bw + (size_t)(n0 + row) * K + k0 + ch * 4);
      *(h4*)&sB[row * 32 + ch * 4] =
          (h4){(_Float16)bv[0], (_Float16)bv[1], (_Float16)bv[2], (_Float16)bv[3]};
    }
    if (kt + 1 < nkt) {  // prefetch next K tile -> global_prefetch_b8
      __builtin_prefetch(A + (size_t)(m0 + (tid >> 1)) * K + k0 + 32 + (tid & 1) * 16, 0, 0);
      __builtin_prefetch(Bw + (size_t)(n0 + (tid >> 2)) * K + k0 + 32 + (tid & 3) * 8, 0, 0);
    }
    __syncthreads();

    // ---- A fragment (16x32, A layout) for this wave ----
    F16x16 af;
    {
      const u32* p = (const u32*)(sA + (wid * 16 + nn) * 32);
#pragma unroll
      for (int i = 0; i < 4; ++i) { af.u[i] = p[hs * 8 + i]; af.u[4 + i] = p[hs * 8 + 4 + i]; }
    }
#pragma unroll
    for (int nt = 0; nt < 4; ++nt) {
      F16x16 bf;
      const u32* p = (const u32*)sB + (nt * 16 + nn) * 16 + hs * 8;
#pragma unroll
      for (int i = 0; i < 8; ++i) bf.u[i] = p[i];
      acc[nt] = __builtin_amdgcn_wmma_f32_16x16x32_f16(
          false, af.v, false, bf.v, (short)0, acc[nt], false, false);
    }
  }
  // ---- epilogue: C layout row = i + hs*8, col = nt*16 + nn ----
#pragma unroll
  for (int nt = 0; nt < 4; ++nt)
#pragma unroll
    for (int i = 0; i < 8; ++i)
      C[(size_t)(m0 + wid * 16 + i + hs * 8) * N + n0 + nt * 16 + nn] = acc[nt][i];
}

// =====================================================================
// RMS-norm + RoPE + gain + f16 convert + relayout to [B,NH,T,HD]
// one wave32 per (b,t,h): lane == head-dim element
// =====================================================================
__global__ void rope_rms_kernel(const float* __restrict__ qkv, const float* __restrict__ gain,
                                const float* __restrict__ cosT, const float* __restrict__ sinT,
                                _Float16* __restrict__ qh, _Float16* __restrict__ kh,
                                _Float16* __restrict__ vh) {
  const int gw   = (blockIdx.x * blockDim.x + threadIdx.x) >> 5;
  const int lane = threadIdx.x & 31;
  const int h = gw % NHH;
  const int t = (gw / NHH) % TT;
  const int b = gw / (NHH * TT);
  const size_t base   = (size_t)(b * TT + t) * DD + h * HDD + lane;
  const size_t stride = (size_t)BT * DD;
  float qv = qkv[base];
  float kv = qkv[base + stride];
  float vv = qkv[base + 2 * stride];
  float sq = qv * qv, sk = kv * kv;
#pragma unroll
  for (int m = 16; m; m >>= 1) { sq += __shfl_xor(sq, m, 32); sk += __shfl_xor(sk, m, 32); }
  float nq = qv * rsqrtf(sq * (1.f / HDD) + 1e-6f);
  float nk = kv * rsqrtf(sk * (1.f / HDD) + 1e-6f);
  float c = cosT[t * (HDD / 2) + (lane & 15)];
  float s = sinT[t * (HDD / 2) + (lane & 15)];
  float pq = __shfl_xor(nq, 16, 32);
  float pk = __shfl_xor(nk, 16, 32);
  float sgn = (lane < 16) ? -1.f : 1.f;   // lo: x1*c - x2*s ; hi: x2*c + x1*s
  float rq = nq * c + pq * s * sgn;
  float rk = nk * c + pk * s * sgn;
  rq *= gain[h] * QK_SCALE;               // fold gain and softmax scale into q
  const size_t ob = ((size_t)(b * NHH + h) * TT + t) * HDD + lane;
  qh[ob] = (_Float16)rq;
  kh[ob] = (_Float16)rk;
  vh[ob] = (_Float16)vv;
}

// =====================================================================
// Flash attention, causal.  grid: (B*NH, T/64), block: 128 = 4 waves.
// wave w owns q rows [by*64 + w*16, +16); key tiles of 32 staged in LDS.
// =====================================================================
__global__ void flash_attn_kernel(const _Float16* __restrict__ qh,
                                  const _Float16* __restrict__ kh,
                                  const _Float16* __restrict__ vh,
                                  _Float16* __restrict__ yh) {
  __shared__ alignas(16) _Float16 sK[32 * 32];      // [key][dim]
  __shared__ alignas(16) _Float16 sV[32 * 32];      // [dim][key] (transposed)
  __shared__ alignas(16) _Float16 sP[4 * 16 * 32];  // per-wave P scratch
  const int tid  = threadIdx.x;
  const int wid  = tid >> 5;
  const int lane = tid & 31;
  const int bh = blockIdx.x;
  const int by = blockIdx.y;
  const int b = bh / NHH, h = bh % NHH;
  const int q0 = by * 64 + wid * 16;
  const int hs = lane >> 4;
  const int nn = lane & 15;

  // Q fragment (A layout) straight from global
  F16x16 qa;
  {
    const u32* p = (const u32*)(qh + ((size_t)bh * TT + q0 + nn) * HDD);
#pragma unroll
    for (int i = 0; i < 4; ++i) { qa.u[i] = p[hs * 8 + i]; qa.u[4 + i] = p[hs * 8 + 4 + i]; }
  }

  const v8f vz = {0, 0, 0, 0, 0, 0, 0, 0};
  v8f o0 = vz, o1 = vz;
  float mstate[8], lstate[8];
#pragma unroll
  for (int i = 0; i < 8; ++i) { mstate[i] = -1e30f; lstate[i] = 0.f; }

  const int nkb = 2 * by + 2;  // key tiles needed by the whole block
  for (int j = 0; j < nkb; ++j) {
    __syncthreads();
    {  // stage K (direct) and V (transposed) tiles: 32 keys x 32 dims
      int row = tid >> 2, c = tid & 3;
      *(uint4*)&sK[row * 32 + c * 8] =
          *(const uint4*)&kh[((size_t)bh * TT + j * 32 + row) * HDD + c * 8];
      union { uint4 q; _Float16 hh[8]; } u;
      u.q = *(const uint4*)&vh[((size_t)bh * TT + j * 32 + row) * HDD + c * 8];
#pragma unroll
      for (int e = 0; e < 8; ++e) sV[(c * 8 + e) * 32 + row] = u.hh[e];
    }
    __syncthreads();
    if (j * 32 > q0 + 15) continue;  // tile fully above diagonal for this wave

    // ---- S = Q @ K^T (two 16x16 subtiles over 32 keys) ----
    F16x16 kb0, kb1;
    {
      const u32* p0 = (const u32*)sK + nn * 16 + hs * 8;
      const u32* p1 = (const u32*)sK + (nn + 16) * 16 + hs * 8;
#pragma unroll
      for (int i = 0; i < 8; ++i) { kb0.u[i] = p0[i]; kb1.u[i] = p1[i]; }
    }
    v8f s0 = __builtin_amdgcn_wmma_f32_16x16x32_f16(false, qa.v, false, kb0.v, (short)0, vz, false, false);
    v8f s1 = __builtin_amdgcn_wmma_f32_16x16x32_f16(false, qa.v, false, kb1.v, (short)0, vz, false, false);

    const bool full = (j * 32 + 31 <= q0);
    if (!full) {
#pragma unroll
      for (int i = 0; i < 8; ++i) {
        int qrow = q0 + i + hs * 8;
        int kc = j * 32 + nn;
        if (kc > qrow)      s0[i] = -1e30f;
        if (kc + 16 > qrow) s1[i] = -1e30f;
      }
    }

    // ---- online softmax (row = (i, lane-half); cols across 16 lanes) ----
    _Float16* sPw = sP + wid * 512;
#pragma unroll
    for (int i = 0; i < 8; ++i) {
      float rmax = fmaxf(s0[i], s1[i]);
#pragma unroll
      for (int m = 8; m; m >>= 1) rmax = fmaxf(rmax, __shfl_xor(rmax, m, 32));
      float mnew = fmaxf(mstate[i], rmax);
      float alpha = __expf(mstate[i] - mnew);
      mstate[i] = mnew;
      float p0 = __expf(s0[i] - mnew);
      float p1 = __expf(s1[i] - mnew);
      float rs = p0 + p1;
#pragma unroll
      for (int m = 8; m; m >>= 1) rs += __shfl_xor(rs, m, 32);
      lstate[i] = lstate[i] * alpha + rs;
      o0[i] *= alpha;
      o1[i] *= alpha;
      sPw[(i + hs * 8) * 32 + nn]      = (_Float16)p0;  // C-layout -> LDS
      sPw[(i + hs * 8) * 32 + nn + 16] = (_Float16)p1;
    }
    asm volatile("s_wait_dscnt 0" ::: "memory");  // P stores visible before A-frag reload

    // ---- O += P @ V ----
    F16x16 pa, vb0, vb1;
    {
      const u32* p = (const u32*)(sPw + nn * 32);
#pragma unroll
      for (int i = 0; i < 4; ++i) { pa.u[i] = p[hs * 8 + i]; pa.u[4 + i] = p[hs * 8 + 4 + i]; }
      const u32* v0 = (const u32*)sV + nn * 16 + hs * 8;
      const u32* v1 = (const u32*)sV + (nn + 16) * 16 + hs * 8;
#pragma unroll
      for (int i = 0; i < 8; ++i) { vb0.u[i] = v0[i]; vb1.u[i] = v1[i]; }
    }
    o0 = __builtin_amdgcn_wmma_f32_16x16x32_f16(false, pa.v, false, vb0.v, (short)0, o0, false, false);
    o1 = __builtin_amdgcn_wmma_f32_16x16x32_f16(false, pa.v, false, vb1.v, (short)0, o1, false, false);
  }

  // ---- epilogue: O/l, f16, scatter back to [B,T,D] head-interleaved ----
#pragma unroll
  for (int i = 0; i < 8; ++i) {
    int qrow = q0 + i + hs * 8;
    float inv = 1.f / lstate[i];
    size_t ob = ((size_t)(b * TT) + qrow) * DD + h * HDD;
    yh[ob + nn]      = (_Float16)(o0[i] * inv);
    yh[ob + nn + 16] = (_Float16)(o1[i] * inv);
  }
}

// =====================================================================
extern "C" void kernel_launch(void* const* d_in, const int* in_sizes, int n_in,
                              void* d_out, int out_size, void* d_ws, size_t ws_size,
                              hipStream_t stream) {
  (void)in_sizes; (void)n_in; (void)out_size; (void)ws_size;
  const float* x     = (const float*)d_in[0];
  const float* Wq    = (const float*)d_in[1];
  const float* Wk    = (const float*)d_in[2];
  const float* Wv    = (const float*)d_in[3];
  const float* Wproj = (const float*)d_in[4];
  const float* gain  = (const float*)d_in[5];
  const float* cosT  = (const float*)d_in[6];
  const float* sinT  = (const float*)d_in[7];

  // workspace carve (~80 MB)
  float* qkv = (float*)d_ws;                         // 3 * 4096*1024 fp32
  const size_t plane = (size_t)BT * DD;
  _Float16* qh = (_Float16*)(qkv + 3 * plane);       // [B,NH,T,HD] f16
  _Float16* kh = qh + plane;
  _Float16* vh = kh + plane;
  _Float16* yh = vh + plane;                         // [B,T,D] f16

  dim3 ggemm(BT / 128, DD / 64);
  gemm_xwT<float><<<ggemm, 256, 0, stream>>>(x, Wq, qkv,             BT, DD, DD);
  gemm_xwT<float><<<ggemm, 256, 0, stream>>>(x, Wk, qkv + plane,     BT, DD, DD);
  gemm_xwT<float><<<ggemm, 256, 0, stream>>>(x, Wv, qkv + 2 * plane, BT, DD, DD);

  rope_rms_kernel<<<(BT * NHH) / 8, 256, 0, stream>>>(qkv, gain, cosT, sinT, qh, kh, vh);

  flash_attn_kernel<<<dim3(BB * NHH, TT / 64), 128, 0, stream>>>(qh, kh, vh, yh);

  gemm_xwT<_Float16><<<ggemm, 256, 0, stream>>>(yh, Wproj, (float*)d_out, BT, DD, DD);
}
